// CSDM_37761352467127
// MI455X (gfx1250) — matrix operand order repaired and checked by reading, not compiled, not run on hardware
//
#include <hip/hip_runtime.h>
#include <math.h>

// ---------------------------------------------------------------------------
// CSDM fused pipeline for MI455X (gfx1250, wave32, WMMA f32_16x16x32_f16).
// Channel-mixing convs on the matrix pipe; LDS operands pre-packed in f16 in
// WMMA fragment order so the inner loops are pure ds_load_b128 -> v_wmma.
// ---------------------------------------------------------------------------

typedef __attribute__((ext_vector_type(16))) _Float16 v16h;
typedef __attribute__((ext_vector_type(8)))  _Float16 v8h;
typedef __attribute__((ext_vector_type(8)))  float    v8f;

#define CCH 32
#define HF  256
#define WF  256
#define HWF 65536

__device__ __forceinline__ float silu_f(float x) { return x / (1.0f + __expf(-x)); }

// A-fragment (16x32 f16): contiguous, element i -> K=2*kA0+i (kA0 = 0 or 8)
__device__ __forceinline__ v16h load_afrag(const _Float16* wrow, int kA0) {
    v8h lo = *(const v8h*)(wrow + 2 * kA0);
    v8h hi = *(const v8h*)(wrow + 2 * kA0 + 8);
    return __builtin_shufflevector(lo, hi, 0, 1, 2, 3, 4, 5, 6, 7,
                                   8, 9, 10, 11, 12, 13, 14, 15);
}

// ---------------------------------------------------------------------------
__global__ void init_kernel(float* p, int n) {
    int i = blockIdx.x * blockDim.x + threadIdx.x;
    if (i < n) p[i] = 0.0f;
}

// ---------------------------------------------------------------------------
// fused avgpool x2 (s1) and x4 (s2): one pass over x
__global__ void __launch_bounds__(256) pool_kernel(const float* __restrict__ x,
                                                   float* __restrict__ s1,
                                                   float* __restrict__ s2) {
    int i  = blockIdx.x * 256 + threadIdx.x;     // over B*C*64*64
    int xs = i & 63;
    int ys = (i >> 6) & 63;
    int bc = i >> 12;                            // b*32+c
    const float* xp = x + (size_t)bc * HWF;
    float tot = 0.0f;
#pragma unroll
    for (int dy = 0; dy < 2; ++dy)
#pragma unroll
        for (int dx = 0; dx < 2; ++dx) {
            int yr = ys * 4 + dy * 2, xc = xs * 4 + dx * 2;
            float v = xp[yr * WF + xc] + xp[yr * WF + xc + 1] +
                      xp[(yr + 1) * WF + xc] + xp[(yr + 1) * WF + xc + 1];
            s1[(size_t)bc * 16384 + (ys * 2 + dy) * 128 + (xs * 2 + dx)] = v * 0.25f;
            tot += v;
        }
    s2[(size_t)bc * 4096 + ys * 64 + xs] = tot * 0.0625f;
}

// ---------------------------------------------------------------------------
// fused ds_block: dwconv3x3 (fp32) + conv1x1(WMMA f16) + BN + SiLU.
// 128 threads = 4 waves; each wave produces 16 pixels x 32 channels.
__global__ void __launch_bounds__(128) ds_block_kernel(
    const float* __restrict__ in, float* __restrict__ out,
    const float* __restrict__ dw, const float* __restrict__ pw,
    const float* __restrict__ g, const float* __restrict__ beta,
    const float* __restrict__ mu, const float* __restrict__ var,
    int H, int W) {
    __shared__ float    tile[CCH * 3 * 68];
    __shared__ float    sdw[CCH * 9];
    __shared__ _Float16 spwh[CCH * CCH];   // f16, [c_out][c_in]
    __shared__ float    ssc[CCH], sbi[CCH];

    const int b   = blockIdx.z;
    const int y   = blockIdx.y;
    const int x0  = blockIdx.x * 64;
    const int tid = threadIdx.x;

    for (int i = tid; i < CCH * 9; i += 128)   sdw[i]  = dw[i];
    for (int i = tid; i < CCH * CCH; i += 128) spwh[i] = (_Float16)pw[i];
    if (tid < CCH) {
        float sc = g[tid] * rsqrtf(var[tid] + 1e-5f);
        ssc[tid] = sc;
        sbi[tid] = beta[tid] - mu[tid] * sc;
    }
    // haloed fp32 input tile, zero-padded (SAME)
    for (int e = tid; e < CCH * 3 * 66; e += 128) {
        int ci = e / (3 * 66);
        int r  = (e / 66) % 3;
        int cc = e % 66;
        int gy = y - 1 + r;
        int gx = x0 - 1 + cc;
        float v = 0.0f;
        if ((unsigned)gy < (unsigned)H && (unsigned)gx < (unsigned)W)
            v = in[(((size_t)b * CCH + ci) * H + gy) * W + gx];
        tile[(ci * 3 + r) * 68 + cc] = v;
    }
    __syncthreads();

    const int lane = tid & 31;
    const int seg  = tid >> 5;
    const int n    = lane & 15;
    const int px   = seg * 16 + n;
    const int kb   = (lane < 16) ? 0 : 16;   // B-matrix K half
    const int m    = lane & 15;
    const int kA0  = (lane < 16) ? 0 : 8;

    // depthwise 3x3 in fp32 -> B fragment (K = c_in, N = pixel)
    v16h bfr;
#pragma unroll
    for (int i = 0; i < 16; ++i) {
        int ci = kb + i;
        const float* t  = &tile[ci * 3 * 68];
        const float* w9 = &sdw[ci * 9];
        float acc = 0.0f;
#pragma unroll
        for (int r = 0; r < 3; ++r)
#pragma unroll
            for (int c = 0; c < 3; ++c)
                acc += t[r * 68 + px + c] * w9[r * 3 + c];
        bfr[i] = (_Float16)acc;
    }

    v16h af0 = load_afrag(&spwh[m * CCH], kA0);
    v16h af1 = load_afrag(&spwh[(m + 16) * CCH], kA0);

    v8f c0 = {}, c1 = {};
    c0 = __builtin_amdgcn_wmma_f32_16x16x32_f16(false, af0, false, bfr, (short)0, c0, false, false);
    c1 = __builtin_amdgcn_wmma_f32_16x16x32_f16(false, af1, false, bfr, (short)0, c1, false, false);

    const int rowAdd = (lane < 16) ? 0 : 8;
#pragma unroll
    for (int r = 0; r < 8; ++r) {
        int co = r + rowAdd;
        float v0 = c0[r] * ssc[co] + sbi[co];
        float v1 = c1[r] * ssc[co + 16] + sbi[co + 16];
        out[(((size_t)b * CCH + co) * H + y) * W + (x0 + px)]      = silu_f(v0);
        out[(((size_t)b * CCH + co + 16) * H + y) * W + (x0 + px)] = silu_f(v1);
    }
}

// ---------------------------------------------------------------------------
// plain conv1x1 (qproj) via WMMA; 256 threads = 8 waves = 128 pixels per row
__global__ void __launch_bounds__(256) conv1x1_kernel(
    const float* __restrict__ in, float* __restrict__ out,
    const float* __restrict__ w, int H, int W) {
    __shared__ _Float16 swh[CCH * CCH];
    const int tid = threadIdx.x;
    for (int i = tid; i < CCH * CCH; i += 256) swh[i] = (_Float16)w[i];
    __syncthreads();

    const int b = blockIdx.z, y = blockIdx.y;
    const int lane = tid & 31, wv = tid >> 5;
    const int n  = lane & 15;
    const int x  = blockIdx.x * 128 + wv * 16 + n;
    const int kb = (lane < 16) ? 0 : 16;

    size_t base = ((size_t)b * CCH) * H * W + (size_t)y * W + x;
    v16h bfr;
#pragma unroll
    for (int i = 0; i < 16; ++i)
        bfr[i] = (_Float16)in[base + (size_t)(kb + i) * H * W];

    const int m = lane & 15;
    const int kA0 = (lane < 16) ? 0 : 8;
    v16h af0 = load_afrag(&swh[m * CCH], kA0);
    v16h af1 = load_afrag(&swh[(m + 16) * CCH], kA0);

    v8f c0 = {}, c1 = {};
    c0 = __builtin_amdgcn_wmma_f32_16x16x32_f16(false, af0, false, bfr, (short)0, c0, false, false);
    c1 = __builtin_amdgcn_wmma_f32_16x16x32_f16(false, af1, false, bfr, (short)0, c1, false, false);

    const int rowAdd = (lane < 16) ? 0 : 8;
#pragma unroll
    for (int r = 0; r < 8; ++r) {
        int co = r + rowAdd;
        out[((size_t)b * CCH + co) * H * W + (size_t)y * W + x]      = c0[r];
        out[((size_t)b * CCH + co + 16) * H * W + (size_t)y * W + x] = c1[r];
    }
}

// ---------------------------------------------------------------------------
// offset head: conv3x3(32->16) as 9 chained WMMAs over f16-packed LDS tile,
// + SiLU + conv1x1(16->12) + bias + softmax over 4 attention channels.
__global__ void __launch_bounds__(128) head_kernel(
    const float* __restrict__ q,
    const float* __restrict__ w1,   // [16][32][3][3]
    const float* __restrict__ w2,   // [12][16]
    const float* __restrict__ b2,   // [12]
    float* __restrict__ off,        // [B][8][H][W]
    float* __restrict__ wsm,        // [B][4][H][W]
    int H, int W) {
    __shared__ _Float16 tileh[3 * 66 * CCH];     // [row][col][ci], f16
    __shared__ _Float16 sw1h[9 * 16 * CCH];      // [tap][m][k], f16
    __shared__ float    sw2[12 * 16];
    __shared__ float    sb2[12];
    __shared__ float    hbuf[16][64];

    const int b   = blockIdx.z;
    const int y   = blockIdx.y;
    const int x0  = blockIdx.x * 64;
    const int tid = threadIdx.x;

    // stage w1 as f16 in fragment-friendly [tap][m][k] order
    for (int e = tid; e < 9 * 16 * CCH; e += 128) {
        int tap = e / (16 * CCH);
        int mm  = (e / CCH) & 15;
        int k   = e & (CCH - 1);
        sw1h[e] = (_Float16)w1[(mm * CCH + k) * 9 + tap];
    }
    for (int i = tid; i < 12 * 16; i += 128) sw2[i] = w2[i];
    if (tid < 12) sb2[tid] = b2[tid];
    // haloed q tile, f16, [row][col][ci] so B fragments are contiguous
    for (int e = tid; e < CCH * 3 * 66; e += 128) {
        int ci = e / (3 * 66);
        int r  = (e / 66) % 3;
        int cc = e % 66;
        int gy = y - 1 + r;
        int gx = x0 - 1 + cc;
        float v = 0.0f;
        if ((unsigned)gy < (unsigned)H && (unsigned)gx < (unsigned)W)
            v = q[(((size_t)b * CCH + ci) * H + gy) * W + gx];
        tileh[(r * 66 + cc) * CCH + ci] = (_Float16)v;
    }
    __syncthreads();

    const int lane = tid & 31;
    const int seg  = tid >> 5;
    const int n    = lane & 15;
    const int px   = seg * 16 + n;
    const int kb   = (lane < 16) ? 0 : 16;
    const int m    = lane & 15;
    const int kA0  = (lane < 16) ? 0 : 8;

    // K=288 GEMM as 9 chained K=32 WMMAs; operands are raw LDS vector loads
    v8f acc = {};
#pragma unroll
    for (int ty = 0; ty < 3; ++ty)
#pragma unroll
        for (int tx = 0; tx < 3; ++tx) {
            int tap = ty * 3 + tx;
            v16h af  = load_afrag(&sw1h[(tap * 16 + m) * CCH], kA0);
            v16h bfr = *(const v16h*)(&tileh[(ty * 66 + px + tx) * CCH + kb]);
            acc = __builtin_amdgcn_wmma_f32_16x16x32_f16(false, af, false, bfr,
                                                         (short)0, acc, false, false);
        }

    const int rowAdd = (lane < 16) ? 0 : 8;
#pragma unroll
    for (int r = 0; r < 8; ++r)
        hbuf[r + rowAdd][px] = silu_f(acc[r]);
    __syncthreads();

    // 16->12 projection + bias + softmax over the 4 attention logits
    if (tid < 64) {
        float o[12];
#pragma unroll
        for (int j = 0; j < 12; ++j) {
            float s = sb2[j];
#pragma unroll
            for (int c = 0; c < 16; ++c) s += sw2[j * 16 + c] * hbuf[c][tid];
            o[j] = s;
        }
        size_t pix = (size_t)y * W + (x0 + tid);
#pragma unroll
        for (int j = 0; j < 8; ++j)
            off[((size_t)b * 8 + j) * H * W + pix] = o[j];
        float mx = fmaxf(fmaxf(o[8], o[9]), fmaxf(o[10], o[11]));
        float e0 = __expf(o[8] - mx), e1 = __expf(o[9] - mx);
        float e2 = __expf(o[10] - mx), e3 = __expf(o[11] - mx);
        float inv = 1.0f / (e0 + e1 + e2 + e3);
        wsm[((size_t)b * 4 + 0) * H * W + pix] = e0 * inv;
        wsm[((size_t)b * 4 + 1) * H * W + pix] = e1 * inv;
        wsm[((size_t)b * 4 + 2) * H * W + pix] = e2 * inv;
        wsm[((size_t)b * 4 + 3) * H * W + pix] = e3 * inv;
    }
}

// ---------------------------------------------------------------------------
// deformable bilinear sampling + weighted combine over K=4, plus per-(b,c)
// partial sums via LDS ds_add_f32 -> one global atomic per channel per block.
__global__ void __launch_bounds__(256) deform_kernel(
    const float* __restrict__ z, const float* __restrict__ off,
    const float* __restrict__ wsm, float* __restrict__ yout,
    float* __restrict__ dsum, int Hc, int Wc) {
    __shared__ float sred[CCH];
    const int tid = threadIdx.x;
    if (tid < CCH) sred[tid] = 0.0f;
    __syncthreads();

    size_t p = (size_t)blockIdx.x * 256 + tid;   // B*H*W; 65536 % 256 == 0
    int b   = (int)(p >> 16);
    int rem = (int)(p & 65535);
    int yy  = rem >> 8, xx = rem & 255;

    float stepx = 2.0f / (float)(Wc - 1), stepy = 2.0f / (float)(Hc - 1);
    float xc = ((float)xx + 0.5f) * ((float)Wc / (float)WF) - 0.5f;
    float yc = ((float)yy + 0.5f) * ((float)Hc / (float)HF) - 0.5f;
    float bgx = 2.0f * xc / (float)(Wc - 1) - 1.0f;
    float bgy = 2.0f * yc / (float)(Hc - 1) - 1.0f;

    float accc[CCH];
#pragma unroll
    for (int c = 0; c < CCH; ++c) accc[c] = 0.0f;

    size_t pix = (size_t)yy * WF + xx;
    const float* zb = z + (size_t)b * CCH * Hc * Wc;
#pragma unroll
    for (int k = 0; k < 4; ++k) {
        float ox = off[((size_t)b * 8 + 2 * k) * HWF + pix];
        float oy = off[((size_t)b * 8 + 2 * k + 1) * HWF + pix];
        float wk = wsm[((size_t)b * 4 + k) * HWF + pix];
        float gx = bgx + tanhf(ox) * (2.0f * stepx);
        float gy = bgy + tanhf(oy) * (2.0f * stepy);
        float ix = fminf(fmaxf((gx + 1.0f) * 0.5f * (Wc - 1), 0.0f), (float)(Wc - 1));
        float iy = fminf(fmaxf((gy + 1.0f) * 0.5f * (Hc - 1), 0.0f), (float)(Hc - 1));
        float x0f = floorf(ix), y0f = floorf(iy);
        float wx = ix - x0f, wy = iy - y0f;
        int x0i = (int)x0f, y0i = (int)y0f;
        int x1i = min(x0i + 1, Wc - 1), y1i = min(y0i + 1, Hc - 1);
        float w00 = wk * (1.0f - wx) * (1.0f - wy);
        float w01 = wk * wx * (1.0f - wy);
        float w10 = wk * (1.0f - wx) * wy;
        float w11 = wk * wx * wy;
        int i00 = y0i * Wc + x0i, i01 = y0i * Wc + x1i;
        int i10 = y1i * Wc + x0i, i11 = y1i * Wc + x1i;
#pragma unroll
        for (int c = 0; c < CCH; ++c) {
            const float* zc = zb + (size_t)c * Hc * Wc;
            accc[c] += w00 * zc[i00] + w01 * zc[i01] + w10 * zc[i10] + w11 * zc[i11];
        }
    }
#pragma unroll
    for (int c = 0; c < CCH; ++c) {
        yout[((size_t)b * CCH + c) * HWF + pix] = accc[c];
        atomicAdd(&sred[c], accc[c]);
    }
    __syncthreads();
    if (tid < CCH) atomicAdd(&dsum[b * CCH + tid], sred[tid]);
}

// ---------------------------------------------------------------------------
// per-(b,c) sum of z0: one block per bin, register accumulate + LDS tree
__global__ void __launch_bounds__(256) mean_kernel(const float* __restrict__ z,
                                                   float* __restrict__ dsum) {
    __shared__ float sred[256];
    const int tid = threadIdx.x;
    const int bin = blockIdx.x;                  // b*32+c, 0..127
    const float* zp = z + (size_t)bin * HWF;
    float s = 0.0f;
    for (int i = tid; i < HWF; i += 256) s += zp[i];
    sred[tid] = s;
    __syncthreads();
    for (int st = 128; st > 0; st >>= 1) {
        if (tid < st) sred[tid] += sred[tid + st];
        __syncthreads();
    }
    if (tid == 0) dsum[bin] = sred[0];
}

// ---------------------------------------------------------------------------
// gate MLP: hid = silu(d @ r_w1 + b1); alpha = softmax(hid @ r_w2 + b2)
__global__ void __launch_bounds__(128) mlp_kernel(
    const float* __restrict__ dsum,  // 384: d0|d1|d2 sums (b*32+c within each)
    const float* __restrict__ w1, const float* __restrict__ b1,
    const float* __restrict__ w2, const float* __restrict__ b2,
    float* __restrict__ alpha) {
    __shared__ float hid[4][32];
    const int tid = threadIdx.x;
    const int b = tid >> 5, l = tid & 31;
    const float inv = 1.0f / (float)HWF;
    float s = b1[l];
    for (int i = 0; i < 96; ++i) {
        float dv = dsum[(i / 32) * 128 + b * 32 + (i % 32)] * inv;
        s += dv * w1[i * 32 + l];
    }
    hid[b][l] = silu_f(s);
    __syncthreads();
    if (tid < 4) {
        float lg[3];
        for (int j = 0; j < 3; ++j) {
            float t = b2[j];
            for (int c = 0; c < 32; ++c) t += hid[tid][c] * w2[c * 3 + j];
            lg[j] = t;
        }
        float mx = fmaxf(lg[0], fmaxf(lg[1], lg[2]));
        float e0 = __expf(lg[0] - mx), e1 = __expf(lg[1] - mx), e2 = __expf(lg[2] - mx);
        float inv2 = 1.0f / (e0 + e1 + e2);
        alpha[tid * 3 + 0] = e0 * inv2;
        alpha[tid * 3 + 1] = e1 * inv2;
        alpha[tid * 3 + 2] = e2 * inv2;
    }
}

// ---------------------------------------------------------------------------
// final: blend = a0*z0 + a1*y1 + a2*y2 ; out = x + conv1x1(blend, final_w)
__global__ void __launch_bounds__(256) final_kernel(
    const float* __restrict__ x, const float* __restrict__ z0,
    const float* __restrict__ y1, const float* __restrict__ y2,
    const float* __restrict__ alpha, const float* __restrict__ fw,
    float* __restrict__ out) {
    __shared__ _Float16 swh[CCH * CCH];
    const int tid = threadIdx.x;
    for (int i = tid; i < CCH * CCH; i += 256) swh[i] = (_Float16)fw[i];
    __syncthreads();

    const int b = blockIdx.z, yy = blockIdx.y;
    const float ga0 = alpha[b * 3 + 0], ga1 = alpha[b * 3 + 1], ga2 = alpha[b * 3 + 2];
    const int lane = tid & 31, wv = tid >> 5;
    const int n  = lane & 15;
    const int xx = blockIdx.x * 128 + wv * 16 + n;
    const int kb = (lane < 16) ? 0 : 16;

    size_t base = ((size_t)b * CCH) * HWF + (size_t)yy * WF + xx;
    v16h bfr;
#pragma unroll
    for (int i = 0; i < 16; ++i) {
        size_t o = base + (size_t)(kb + i) * HWF;
        bfr[i] = (_Float16)(ga0 * z0[o] + ga1 * y1[o] + ga2 * y2[o]);
    }
    const int m = lane & 15;
    const int kA0 = (lane < 16) ? 0 : 8;
    v16h af0 = load_afrag(&swh[m * CCH], kA0);
    v16h af1 = load_afrag(&swh[(m + 16) * CCH], kA0);

    v8f c0 = {}, c1 = {};
    c0 = __builtin_amdgcn_wmma_f32_16x16x32_f16(false, af0, false, bfr, (short)0, c0, false, false);
    c1 = __builtin_amdgcn_wmma_f32_16x16x32_f16(false, af1, false, bfr, (short)0, c1, false, false);

    const int rowAdd = (lane < 16) ? 0 : 8;
#pragma unroll
    for (int r = 0; r < 8; ++r) {
        int co = r + rowAdd;
        size_t o0 = ((size_t)b * CCH + co) * HWF + (size_t)yy * WF + xx;
        size_t o1 = ((size_t)b * CCH + co + 16) * HWF + (size_t)yy * WF + xx;
        out[o0] = x[o0] + c0[r];
        out[o1] = x[o1] + c1[r];
    }
}

// ---------------------------------------------------------------------------
extern "C" void kernel_launch(void* const* d_in, const int* in_sizes, int n_in,
                              void* d_out, int out_size, void* d_ws, size_t ws_size,
                              hipStream_t stream) {
    (void)in_sizes; (void)n_in; (void)out_size; (void)ws_size;

    const float* x       = (const float*)d_in[0];
    const float* ds_dw[3] = {(const float*)d_in[1], (const float*)d_in[7],  (const float*)d_in[13]};
    const float* ds_pw[3] = {(const float*)d_in[2], (const float*)d_in[8],  (const float*)d_in[14]};
    const float* ds_g[3]  = {(const float*)d_in[3], (const float*)d_in[9],  (const float*)d_in[15]};
    const float* ds_b[3]  = {(const float*)d_in[4], (const float*)d_in[10], (const float*)d_in[16]};
    const float* ds_m[3]  = {(const float*)d_in[5], (const float*)d_in[11], (const float*)d_in[17]};
    const float* ds_v[3]  = {(const float*)d_in[6], (const float*)d_in[12], (const float*)d_in[18]};
    const float* qproj_w = (const float*)d_in[19];
    const float* h1_w1 = (const float*)d_in[20];
    const float* h1_w2 = (const float*)d_in[21];
    const float* h1_b2 = (const float*)d_in[22];
    const float* h2_w1 = (const float*)d_in[23];
    const float* h2_w2 = (const float*)d_in[24];
    const float* h2_b2 = (const float*)d_in[25];
    const float* r_w1 = (const float*)d_in[26];
    const float* r_b1 = (const float*)d_in[27];
    const float* r_w2 = (const float*)d_in[28];
    const float* r_b2 = (const float*)d_in[29];
    const float* final_w = (const float*)d_in[30];
    float* out = (float*)d_out;

    // workspace layout (floats)
    float* ws = (float*)d_ws;
    size_t o = 0;
    float* W_s1   = ws + o; o += (size_t)4 * CCH * 128 * 128;
    float* W_s2   = ws + o; o += (size_t)4 * CCH * 64 * 64;
    float* W_z0   = ws + o; o += (size_t)4 * CCH * HWF;
    float* W_z1   = ws + o; o += (size_t)4 * CCH * 128 * 128;
    float* W_z2   = ws + o; o += (size_t)4 * CCH * 64 * 64;
    float* W_q    = ws + o; o += (size_t)4 * CCH * HWF;
    float* W_off1 = ws + o; o += (size_t)4 * 8 * HWF;
    float* W_w1s  = ws + o; o += (size_t)4 * 4 * HWF;
    float* W_off2 = ws + o; o += (size_t)4 * 8 * HWF;
    float* W_w2s  = ws + o; o += (size_t)4 * 4 * HWF;
    float* W_y1   = ws + o; o += (size_t)4 * CCH * HWF;
    float* W_y2   = ws + o; o += (size_t)4 * CCH * HWF;
    float* W_dsum = ws + o; o += 384;   // d0|d1|d2 sums
    float* W_alph = ws + o; o += 12;

    init_kernel<<<2, 256, 0, stream>>>(W_dsum, 396);        // zero dsum + alpha
    pool_kernel<<<2048, 256, 0, stream>>>(x, W_s1, W_s2);

    ds_block_kernel<<<dim3(4, 256, 4), 128, 0, stream>>>(
        x, W_z0, ds_dw[0], ds_pw[0], ds_g[0], ds_b[0], ds_m[0], ds_v[0], 256, 256);
    ds_block_kernel<<<dim3(2, 128, 4), 128, 0, stream>>>(
        W_s1, W_z1, ds_dw[1], ds_pw[1], ds_g[1], ds_b[1], ds_m[1], ds_v[1], 128, 128);
    ds_block_kernel<<<dim3(1, 64, 4), 128, 0, stream>>>(
        W_s2, W_z2, ds_dw[2], ds_pw[2], ds_g[2], ds_b[2], ds_m[2], ds_v[2], 64, 64);

    conv1x1_kernel<<<dim3(2, 256, 4), 256, 0, stream>>>(W_z0, W_q, qproj_w, 256, 256);

    head_kernel<<<dim3(4, 256, 4), 128, 0, stream>>>(W_q, h1_w1, h1_w2, h1_b2,
                                                     W_off1, W_w1s, 256, 256);
    head_kernel<<<dim3(4, 256, 4), 128, 0, stream>>>(W_q, h2_w1, h2_w2, h2_b2,
                                                     W_off2, W_w2s, 256, 256);

    mean_kernel<<<128, 256, 0, stream>>>(W_z0, W_dsum);                      // d0
    deform_kernel<<<1024, 256, 0, stream>>>(W_z1, W_off1, W_w1s, W_y1,
                                            W_dsum + 128, 128, 128);         // y1,d1
    deform_kernel<<<1024, 256, 0, stream>>>(W_z2, W_off2, W_w2s, W_y2,
                                            W_dsum + 256, 64, 64);           // y2,d2

    mlp_kernel<<<1, 128, 0, stream>>>(W_dsum, r_w1, r_b1, r_w2, r_b2, W_alph);

    final_kernel<<<dim3(2, 256, 4), 256, 0, stream>>>(x, W_z0, W_y1, W_y2,
                                                      W_alph, final_w, out);
}
